// MultiHeadAttentionLayer_48868137894035
// MI455X (gfx1250) — matrix operand order repaired and checked by reading, not compiled, not run on hardware
//
#include <hip/hip_runtime.h>

typedef __attribute__((ext_vector_type(16))) _Float16 v16h;
typedef __attribute__((ext_vector_type(8)))  float    v8f;

#define HIDDEN  1024
#define NHEADS  16
#define HEADDIM 64
#define BATCH   4
#define SEQ     2048
#define MTOT    (BATCH * SEQ)   /* 8192 */
#define NEGVAL  (-1e10f)

// ---------------------------------------------------------------------------
// WMMA fragment loaders (wave32; contiguous k order k = 16*(lane/16) + element)
// ---------------------------------------------------------------------------

// A fragment 16x32 (MxK), f16, from row-major f32 source (row stride ld).
// lane<16: K = k0 + 0..15;  lane>=16: K = k0 + 16..31.
__device__ __forceinline__ v16h load_a_f32(const float* __restrict__ src,
                                           int row0, int ld, int k0, int lane) {
  const int r  = row0 + (lane & 15);
  const int kb = k0 + ((lane & 16) ? 16 : 0);
  const float* p = src + (size_t)r * ld + kb;
  v16h a;
#pragma unroll
  for (int e = 0; e < 8; ++e) a[e] = (_Float16)p[e];
#pragma unroll
  for (int e = 0; e < 8; ++e) a[8 + e] = (_Float16)p[8 + e];
  return a;
}

// A fragment 16x32 from row-major f16 source.
__device__ __forceinline__ v16h load_a_f16(const _Float16* __restrict__ src,
                                           int row0, int ld, int k0, int lane) {
  const int r  = row0 + (lane & 15);
  const int kb = k0 + ((lane & 16) ? 16 : 0);
  const _Float16* p = src + (size_t)r * ld + kb;
  v16h a;
#pragma unroll
  for (int e = 0; e < 8; ++e) a[e] = p[e];
#pragma unroll
  for (int e = 0; e < 8; ++e) a[8 + e] = p[8 + e];
  return a;
}

// B fragment 32x16 (KxN) where B(k,n) = W[n*ld + k]   (i.e. X @ W^T), f32 src.
// lane<16: K = k0+0..15;  lane>=16: K = k0+16..31;  N = n0 + (lane&15).
__device__ __forceinline__ v16h load_bT_f32(const float* __restrict__ W,
                                            int n0, int ld, int k0, int lane) {
  const int n  = n0 + (lane & 15);
  const int kb = k0 + ((lane & 16) ? 16 : 0);
  const float* p = W + (size_t)n * ld + kb;
  v16h b;
#pragma unroll
  for (int e = 0; e < 16; ++e) b[e] = (_Float16)p[e];
  return b;
}

// B fragment 32x16 where B(k,n) = src[n*ld + k], f16 source (K^T case).
__device__ __forceinline__ v16h load_bT_f16(const _Float16* __restrict__ src,
                                            int n0, int ld, int k0, int lane) {
  const int n  = n0 + (lane & 15);
  const int kb = k0 + ((lane & 16) ? 16 : 0);
  const _Float16* p = src + (size_t)n * ld + kb;
  v16h b;
#pragma unroll
  for (int e = 0; e < 16; ++e) b[e] = p[e];
  return b;
}

// B fragment 32x16 where B(k,n) = src[k*ld + n], f16 source (V case).
__device__ __forceinline__ v16h load_bRM_f16(const _Float16* __restrict__ src,
                                             int n0, int ld, int k0, int lane) {
  const int n  = n0 + (lane & 15);
  const int kb = k0 + ((lane & 16) ? 16 : 0);
  v16h b;
#pragma unroll
  for (int e = 0; e < 16; ++e) b[e] = src[(size_t)(kb + e) * ld + n];
  return b;
}

#define WMMA_F16(a, b, c) \
  __builtin_amdgcn_wmma_f32_16x16x32_f16(false, (a), false, (b), (short)0, (c), false, false)

// ---------------------------------------------------------------------------
// 1) QKV projection:  dst[b,h,s,d] (f16) = (X @ W^T + bias)  split-head
//    M=8192, N=1024, K=1024.  Block = 8 waves, 128x64 tile; wave = 32x32.
// ---------------------------------------------------------------------------
__global__ __launch_bounds__(256) void mha_proj_qkv(const float* __restrict__ X,
                                                    const float* __restrict__ W,
                                                    const float* __restrict__ bias,
                                                    _Float16* __restrict__ dst) {
  const int lane = threadIdx.x & 31;
  const int wave = threadIdx.x >> 5;
  const int wm = wave >> 1, wn = wave & 1;
  const int m0 = blockIdx.x * 128 + wm * 32;
  const int n0 = blockIdx.y * 64 + wn * 32;

  v8f acc[2][2] = {};
#pragma unroll 1
  for (int k0 = 0; k0 < HIDDEN; k0 += 32) {
    if (k0 + 32 < HIDDEN) {  // gfx1250 global_prefetch_b8 of next K-tile
      __builtin_prefetch(X + (size_t)(m0 + (lane & 15)) * HIDDEN + k0 + 32, 0, 1);
      __builtin_prefetch(W + (size_t)(n0 + (lane & 15)) * HIDDEN + k0 + 32, 0, 1);
    }
    v16h a0 = load_a_f32(X, m0,      HIDDEN, k0, lane);
    v16h a1 = load_a_f32(X, m0 + 16, HIDDEN, k0, lane);
    v16h b0 = load_bT_f32(W, n0,      HIDDEN, k0, lane);
    v16h b1 = load_bT_f32(W, n0 + 16, HIDDEN, k0, lane);
    acc[0][0] = WMMA_F16(a0, b0, acc[0][0]);
    acc[0][1] = WMMA_F16(a0, b1, acc[0][1]);
    acc[1][0] = WMMA_F16(a1, b0, acc[1][0]);
    acc[1][1] = WMMA_F16(a1, b1, acc[1][1]);
  }

  const int rbase = (lane & 16) ? 8 : 0;
#pragma unroll
  for (int i = 0; i < 2; ++i)
#pragma unroll
    for (int j = 0; j < 2; ++j) {
      const int n = n0 + j * 16 + (lane & 15);
      const float bv = bias[n];
      const int h = n >> 6, d = n & 63;
#pragma unroll
      for (int r = 0; r < 8; ++r) {
        const int m = m0 + i * 16 + rbase + r;
        const int b = m >> 11, s = m & (SEQ - 1);
        dst[(((size_t)b * NHEADS + h) * SEQ + s) * HEADDIM + d] =
            (_Float16)(acc[i][j][r] + bv);
      }
    }
}

// ---------------------------------------------------------------------------
// 2) energy[bh,m,n] = (Q K^T)/8, masked.  Per (b,h): M=N=2048, K=64.
// ---------------------------------------------------------------------------
__global__ __launch_bounds__(256) void mha_energy(const _Float16* __restrict__ qh,
                                                  const _Float16* __restrict__ kh,
                                                  const int* __restrict__ mask,
                                                  float* __restrict__ attn) {
  const int lane = threadIdx.x & 31;
  const int wave = threadIdx.x >> 5;
  const int wm = wave >> 1, wn = wave & 1;
  const int bh = blockIdx.z;
  const int b  = bh >> 4;
  const int m0 = blockIdx.x * 128 + wm * 32;
  const int n0 = blockIdx.y * 64 + wn * 32;

  const _Float16* q = qh + (size_t)bh * SEQ * HEADDIM;
  const _Float16* k = kh + (size_t)bh * SEQ * HEADDIM;

  v8f acc[2][2] = {};
#pragma unroll
  for (int k0 = 0; k0 < HEADDIM; k0 += 32) {
    v16h a0 = load_a_f16(q, m0,      HEADDIM, k0, lane);
    v16h a1 = load_a_f16(q, m0 + 16, HEADDIM, k0, lane);
    v16h b0 = load_bT_f16(k, n0,      HEADDIM, k0, lane);
    v16h b1 = load_bT_f16(k, n0 + 16, HEADDIM, k0, lane);
    acc[0][0] = WMMA_F16(a0, b0, acc[0][0]);
    acc[0][1] = WMMA_F16(a0, b1, acc[0][1]);
    acc[1][0] = WMMA_F16(a1, b0, acc[1][0]);
    acc[1][1] = WMMA_F16(a1, b1, acc[1][1]);
  }

  float* E = attn + (size_t)bh * SEQ * SEQ;
  const int rbase = (lane & 16) ? 8 : 0;
#pragma unroll
  for (int i = 0; i < 2; ++i)
#pragma unroll
    for (int j = 0; j < 2; ++j) {
      const int n = n0 + j * 16 + (lane & 15);
      const int mv = mask[(size_t)b * SEQ + n];
#pragma unroll
      for (int r = 0; r < 8; ++r) {
        const int m = m0 + i * 16 + rbase + r;
        float e = acc[i][j][r] * 0.125f;              // 1/sqrt(64)
        if (mv == 0) e = NEGVAL;
        E[(size_t)m * SEQ + n] = e;
      }
    }
}

// ---------------------------------------------------------------------------
// 3) FUSED softmax + (A @ V).  One block owns a 128-row x 2048-col attention
//    tile of one (b,h): row stats -> in-place normalize (this write IS the
//    returned attention tensor) -> WMMA A@V from the L2-resident tile.
//    Per (b,h): M=2048, N=64, K=2048.
// ---------------------------------------------------------------------------
__global__ __launch_bounds__(256) void mha_softmax_av(float* __restrict__ attn,
                                                      const _Float16* __restrict__ vh,
                                                      _Float16* __restrict__ ctx) {
  __shared__ float red[256];
  __shared__ float rowmax[128];
  __shared__ float rowinv[128];

  const int tid  = threadIdx.x;
  const int lane = tid & 31;
  const int wave = tid >> 5;
  const int wm = wave >> 1, wn = wave & 1;
  const int bh = blockIdx.z;
  const int b  = bh >> 4, h = bh & 15;
  const int blkrow = blockIdx.x * 128;

  float*          A = attn + (size_t)bh * SEQ * SEQ;   // full (b,h) slice
  float*          E = A + (size_t)blkrow * SEQ;        // this block's 128 rows
  const _Float16* V = vh + (size_t)bh * SEQ * HEADDIM;

  // --- phase 1: per-row max and sum(exp), 2 threads per row ---------------
  {
    const int rr = tid >> 1, half = tid & 1;
    const float* rp = E + (size_t)rr * SEQ + half * (SEQ / 2);
    float mx = -3.4e38f;
    for (int i = 0; i < SEQ / 2; ++i) mx = fmaxf(mx, rp[i]);
    red[tid] = mx;
    __syncthreads();
    const float rm = fmaxf(red[rr * 2], red[rr * 2 + 1]);
    __syncthreads();
    float s = 0.f;
    for (int i = 0; i < SEQ / 2; ++i) s += __expf(rp[i] - rm);
    red[tid] = s;
    __syncthreads();
    if (half == 0) {
      rowmax[rr] = rm;
      rowinv[rr] = 1.0f / (red[rr * 2] + red[rr * 2 + 1]);
    }
    __syncthreads();
  }

  // --- phase 2: normalize in place, coalesced (cols across threads) -------
  for (int idx = tid; idx < 128 * SEQ; idx += 256) {
    const int r = idx >> 11;            // SEQ == 2048
    float* p = E + idx;                 // row-major contiguous block
    *p = __expf(*p - rowmax[r]) * rowinv[r];
  }
  __syncthreads();

  // --- phase 3: WMMA  ctx = A_norm @ V  (A tile is L2-resident) -----------
  const int m0 = blkrow + wm * 32;
  const int n0 = wn * 32;               // N = 64 total

  v8f acc[2][2] = {};
#pragma unroll 1
  for (int k0 = 0; k0 < SEQ; k0 += 32) {
    if (k0 + 32 < SEQ)
      __builtin_prefetch(A + (size_t)(m0 + (lane & 15)) * SEQ + k0 + 32, 0, 1);
    v16h a0 = load_a_f32(A, m0,      SEQ, k0, lane);
    v16h a1 = load_a_f32(A, m0 + 16, SEQ, k0, lane);
    v16h b0 = load_bRM_f16(V, n0,      HEADDIM, k0, lane);
    v16h b1 = load_bRM_f16(V, n0 + 16, HEADDIM, k0, lane);
    acc[0][0] = WMMA_F16(a0, b0, acc[0][0]);
    acc[0][1] = WMMA_F16(a0, b1, acc[0][1]);
    acc[1][0] = WMMA_F16(a1, b0, acc[1][0]);
    acc[1][1] = WMMA_F16(a1, b1, acc[1][1]);
  }

  const int rbase = (lane & 16) ? 8 : 0;
#pragma unroll
  for (int i = 0; i < 2; ++i)
#pragma unroll
    for (int j = 0; j < 2; ++j) {
      const int d = n0 + j * 16 + (lane & 15);
#pragma unroll
      for (int r = 0; r < 8; ++r) {
        const int s = m0 + i * 16 + rbase + r;
        ctx[((size_t)b * SEQ + s) * HIDDEN + h * HEADDIM + d] =
            (_Float16)acc[i][j][r];
      }
    }
}

// ---------------------------------------------------------------------------
// 4) out = ctx @ Wo^T + bo.  M=8192, N=1024, K=1024, f32 output.
// ---------------------------------------------------------------------------
__global__ __launch_bounds__(256) void mha_out_proj(const _Float16* __restrict__ ctx,
                                                    const float* __restrict__ Wo,
                                                    const float* __restrict__ bo,
                                                    float* __restrict__ out) {
  const int lane = threadIdx.x & 31;
  const int wave = threadIdx.x >> 5;
  const int wm = wave >> 1, wn = wave & 1;
  const int m0 = blockIdx.x * 128 + wm * 32;
  const int n0 = blockIdx.y * 64 + wn * 32;

  v8f acc[2][2] = {};
#pragma unroll 1
  for (int k0 = 0; k0 < HIDDEN; k0 += 32) {
    if (k0 + 32 < HIDDEN)
      __builtin_prefetch(Wo + (size_t)(n0 + (lane & 15)) * HIDDEN + k0 + 32, 0, 1);
    v16h a0 = load_a_f16(ctx, m0,      HIDDEN, k0, lane);
    v16h a1 = load_a_f16(ctx, m0 + 16, HIDDEN, k0, lane);
    v16h b0 = load_bT_f32(Wo, n0,      HIDDEN, k0, lane);
    v16h b1 = load_bT_f32(Wo, n0 + 16, HIDDEN, k0, lane);
    acc[0][0] = WMMA_F16(a0, b0, acc[0][0]);
    acc[0][1] = WMMA_F16(a0, b1, acc[0][1]);
    acc[1][0] = WMMA_F16(a1, b0, acc[1][0]);
    acc[1][1] = WMMA_F16(a1, b1, acc[1][1]);
  }

  const int rbase = (lane & 16) ? 8 : 0;
#pragma unroll
  for (int i = 0; i < 2; ++i)
#pragma unroll
    for (int j = 0; j < 2; ++j) {
      const int n = n0 + j * 16 + (lane & 15);
      const float bv = bo[n];
#pragma unroll
      for (int r = 0; r < 8; ++r) {
        const int m = m0 + i * 16 + rbase + r;
        out[(size_t)m * HIDDEN + n] = acc[i][j][r] + bv;
      }
    }
}

// ---------------------------------------------------------------------------
extern "C" void kernel_launch(void* const* d_in, const int* in_sizes, int n_in,
                              void* d_out, int out_size, void* d_ws, size_t ws_size,
                              hipStream_t stream) {
  const float* query = (const float*)d_in[0];
  const float* keyx  = (const float*)d_in[1];
  const float* value = (const float*)d_in[2];
  const int*   mask  = (const int*)d_in[3];
  const float* Wq = (const float*)d_in[4];
  const float* bq = (const float*)d_in[5];
  const float* Wk = (const float*)d_in[6];
  const float* bk = (const float*)d_in[7];
  const float* Wv = (const float*)d_in[8];
  const float* bv = (const float*)d_in[9];
  const float* Wo = (const float*)d_in[10];
  const float* bo = (const float*)d_in[11];

  float* out  = (float*)d_out;
  float* attn = out + (size_t)BATCH * SEQ * HIDDEN;   // tuple: (out, attention)

  const size_t ELEMS = (size_t)BATCH * SEQ * HIDDEN;  // 8,388,608
  _Float16* qh  = (_Float16*)d_ws;
  _Float16* kh  = qh + ELEMS;
  _Float16* vh  = kh + ELEMS;
  _Float16* ctx = vh + ELEMS;                         // 64 MB total f16 scratch

  const dim3 blk(256);

  // 1) fused-layout QKV projections (f32 in, f16 split-head out)
  const dim3 gproj(MTOT / 128, HIDDEN / 64, 1);
  hipLaunchKernelGGL(mha_proj_qkv, gproj, blk, 0, stream, query, Wq, bq, qh);
  hipLaunchKernelGGL(mha_proj_qkv, gproj, blk, 0, stream, keyx,  Wk, bk, kh);
  hipLaunchKernelGGL(mha_proj_qkv, gproj, blk, 0, stream, value, Wv, bv, vh);

  // 2) energy = Q K^T / sqrt(d) + mask  (raw scores into attention buffer)
  const dim3 gen(SEQ / 128, SEQ / 64, BATCH * NHEADS);
  hipLaunchKernelGGL(mha_energy, gen, blk, 0, stream, qh, kh, mask, attn);

  // 3) fused softmax (in-place -> final attention output) + ctx = A @ V
  const dim3 gav(SEQ / 128, 1, BATCH * NHEADS);
  hipLaunchKernelGGL(mha_softmax_av, gav, blk, 0, stream, attn, vh, ctx);

  // 4) out = ctx @ Wo^T + bo
  hipLaunchKernelGGL(mha_out_proj, gproj, blk, 0, stream, ctx, Wo, bo, out);
}
